// Spatial_33775622816248
// MI455X (gfx1250) — matrix operand order repaired and checked by reading, not compiled, not run on hardware
//
#include <hip/hip_runtime.h>

// ---------------------------------------------------------------------------
// Shapes: BS=4, CLOSE=8, N=8192, K=16, D=64, H=4, FF=256, DH=16
// Outputs: sq_c (4,8192,8) then tgt_out (4,8,8192), both f32, concat in d_out.
// ---------------------------------------------------------------------------

#define BSZ   4
#define NPTS  8192
#define CL    8
#define KNN   16
#define DIM   64
#define NH    4
#define DH    16
#define FF    256

typedef __attribute__((ext_vector_type(16))) _Float16 v16h;
typedef __attribute__((ext_vector_type(8)))  float    v8f;
typedef __attribute__((ext_vector_type(8)))  _Float16 h8;
typedef __attribute__((ext_vector_type(4)))  float    f32x4;
typedef __attribute__((ext_vector_type(4)))  unsigned int u32x4;
typedef __attribute__((ext_vector_type(8)))  unsigned int u32x8;

// Packed-weight blob: B fragments in fragment-major layout.
// One tile = one 32x16 f16 B fragment = 32 lanes x 16 f16 (32 B) = 1 KB.
#define T_EMB 0    //  4 tiles : Wemb  (K= 8 pad32, N=64)
#define T_WQ  4    //  8 tiles : Wq    (K=64, N=64)
#define T_WK  12   //  8 tiles : Wk
#define T_WV  20   //  8 tiles : Wv
#define T_WO  28   //  8 tiles : Wo
#define T_FF1 36   // 32 tiles : Wff1  (K=64, N=256)
#define T_FF2 68   // 32 tiles : Wff2  (K=256, N=64)
#define T_OUT 100  //  2 tiles : Wout  (K=64, N=8 pad16)
#define NTILES 102
// Blob = NTILES*512 f16 = 104448 B = 26112 dwords = 1632 x 16 dwords (2D TDM)
#define WBLOB_DW_X 1632u
#define WBLOB_DW_Y 16u

// ---------------------------------------------------------------------------
// CDNA5 WMMA fragment helpers (wave32, v_wmma_f32_16x16x32_f16)
//   A: 16x32 f16. lane: m=lane&15, half=lane>>4.
//      VGPR v(0..3): K = half*8 + 2v,2v+1 ; VGPR v(4..7): K = 16 + half*8 + ...
//   B: 32x16 f16. lane: n=lane&15, half=lane>>4. K as A: v(0..3) half*8 + 2v,2v+1; v(4..7) 16 + half*8 + ...
//   C/D: 16x16 f32. lane: n=lane&15; VGPR v holds row m = v + 8*(lane>>4)
// ---------------------------------------------------------------------------

// A fragment from f32 LDS (16 rows x lda), always fully in-bounds:
// 4x ds_load_b128 + packed f32->f16 converts. (K-padded matrices are stored
// zero-filled so no guarded path is ever needed.)
__device__ __forceinline__ v16h frag_a(const float* a, int lda, int k0,
                                       int lane) {
  int m = lane & 15, hf = lane >> 4;
  const float* r0 = a + m * lda + k0 + hf * 8;  // 16B aligned: lda in {32,64,256}
  f32x4 p0 = *(const f32x4*)(r0);
  f32x4 p1 = *(const f32x4*)(r0 + 4);
  f32x4 p2 = *(const f32x4*)(r0 + 16);
  f32x4 p3 = *(const f32x4*)(r0 + 20);
  v16h out;
#pragma unroll
  for (int j = 0; j < 4; ++j) {
    out[j]      = (_Float16)p0[j];
    out[4 + j]  = (_Float16)p1[j];
    out[8 + j]  = (_Float16)p2[j];
    out[12 + j] = (_Float16)p3[j];
  }
  return out;
}

// B fragment from the fragment-packed LDS blob: one 32B contiguous load.
__device__ __forceinline__ v16h frag_b_packed(const _Float16* blob, int tile,
                                              int lane) {
  return ((const v16h*)(blob + tile * 512))[lane];
}

// One 16x16 D tile: D[:, n0:n0+16] = A(16 x kdim) * W using packed B tiles.
__device__ __forceinline__ void gemm_tile(const float* A, int lda, int kdim,
                                          const _Float16* blob, int tbase,
                                          int ntn, int n0, float* D, int ldd,
                                          int lane, bool relu) {
  v8f c = {};
  for (int k0 = 0; k0 < kdim; k0 += 32) {
    v16h a = frag_a(A, lda, k0, lane);
    v16h b = frag_b_packed(blob, tbase + (k0 >> 5) * ntn + (n0 >> 4), lane);
    c = __builtin_amdgcn_wmma_f32_16x16x32_f16(false, a, false, b, (short)0, c,
                                               false, false);
  }
  int n = n0 + (lane & 15);
  int mo = (lane >> 4) * 8;
#pragma unroll
  for (int v = 0; v < 8; ++v) {
    float x = c[v];
    if (relu) x = fmaxf(x, 0.0f);
    D[(v + mo) * ldd + n] = x;
  }
}

// ---------------------------------------------------------------------------
// Kernel 0: transpose x_c -> x0, normalize -> xn, emit tgt_out output.
// ---------------------------------------------------------------------------
__global__ void prep_kernel(const float* __restrict__ xc, float* __restrict__ x0,
                            float* __restrict__ xn, float* __restrict__ out) {
  int gid = blockIdx.x * blockDim.x + threadIdx.x;  // 0 .. 32767
  int b = gid >> 13;
  int n = gid & (NPTS - 1);
  float v[CL];
  float ss = 0.0f;
#pragma unroll
  for (int c = 0; c < CL; ++c) {
    v[c] = xc[((size_t)(b * CL + c) * 2) * NPTS + n];  // plane i=0
    ss += v[c] * v[c];
  }
  float inv = 1.0f / (sqrtf(ss) + 1e-12f);
#pragma unroll
  for (int c = 0; c < CL; ++c) {
    x0[(size_t)gid * CL + c] = v[c];
    xn[(size_t)gid * CL + c] = v[c] * inv;
    out[(size_t)BSZ * NPTS * CL + (size_t)(b * CL + c) * NPTS + n] = v[c];
  }
}

// ---------------------------------------------------------------------------
// Kernel 1: pack all weights (f32 -> f16) into fragment-major B-tile blob.
// ---------------------------------------------------------------------------
__global__ void pack_kernel(const float* __restrict__ Wemb,
                            const float* __restrict__ Wqkv,
                            const float* __restrict__ Wo,
                            const float* __restrict__ Wff1,
                            const float* __restrict__ Wff2,
                            const float* __restrict__ Wout,
                            _Float16* __restrict__ blob) {
  int tile = blockIdx.x;
  const float* W; int ldw, kact, nact, k0, n0;
  if (tile < T_WQ)        { int t = tile - T_EMB; W = Wemb;               ldw = 64;  kact = 8;   nact = 64;  k0 = 0;            n0 = t * 16; }
  else if (tile < T_WK)   { int t = tile - T_WQ;  W = Wqkv;               ldw = 64;  kact = 64;  nact = 64;  k0 = (t >> 2) * 32; n0 = (t & 3) * 16; }
  else if (tile < T_WV)   { int t = tile - T_WK;  W = Wqkv + DIM * DIM;   ldw = 64;  kact = 64;  nact = 64;  k0 = (t >> 2) * 32; n0 = (t & 3) * 16; }
  else if (tile < T_WO)   { int t = tile - T_WV;  W = Wqkv + 2 * DIM * DIM; ldw = 64; kact = 64; nact = 64;  k0 = (t >> 2) * 32; n0 = (t & 3) * 16; }
  else if (tile < T_FF1)  { int t = tile - T_WO;  W = Wo;                 ldw = 64;  kact = 64;  nact = 64;  k0 = (t >> 2) * 32; n0 = (t & 3) * 16; }
  else if (tile < T_FF2)  { int t = tile - T_FF1; W = Wff1;               ldw = 256; kact = 64;  nact = 256; k0 = (t >> 4) * 32; n0 = (t & 15) * 16; }
  else if (tile < T_OUT)  { int t = tile - T_FF2; W = Wff2;               ldw = 64;  kact = 256; nact = 64;  k0 = (t >> 2) * 32; n0 = (t & 3) * 16; }
  else                    { int t = tile - T_OUT; W = Wout;               ldw = 8;   kact = 64;  nact = 8;   k0 = t * 32;        n0 = 0; }
  for (int i = threadIdx.x; i < 512; i += 256) {
    int lane = i >> 4, e = i & 15;
    int k = k0 + (lane >> 4) * 8 + e + 8 * (e >> 3);
    int n = n0 + (lane & 15);
    float x = (k < kact && n < nact) ? W[k * ldw + n] : 0.0f;
    blob[tile * 512 + i] = (_Float16)x;
  }
}

// ---------------------------------------------------------------------------
// Kernel 2: cosine-similarity top-16 per row. 256 threads = 256 rows; whole
// batch staged in LDS as f16 (128 KB); register-resident top-16 per thread.
// ---------------------------------------------------------------------------
__global__ void topk_kernel(const float* __restrict__ xn, int* __restrict__ idx) {
  __shared__ __align__(16) _Float16 s_xn[NPTS * CL];  // 128 KB
  int b = blockIdx.x >> 5;
  int chunk = blockIdx.x & 31;
  int tid = threadIdx.x;
  const float* xb = xn + (size_t)b * NPTS * CL;

  for (int i = tid; i < NPTS * CL; i += 256) s_xn[i] = (_Float16)xb[i];
  __syncthreads();

  int r = chunk * 256 + tid;
  float rw[CL];
#pragma unroll
  for (int c = 0; c < CL; ++c) rw[c] = xb[(size_t)r * CL + c];

  float tv[KNN];
  int ti[KNN];
#pragma unroll
  for (int s = 0; s < KNN; ++s) { tv[s] = -1e30f; ti[s] = 0; }

  const h8* col = (const h8*)s_xn;
  for (int j = 0; j < NPTS; ++j) {
    h8 vv = col[j];
    float d = 0.0f;
#pragma unroll
    for (int c = 0; c < CL; ++c) d += (float)vv[c] * rw[c];
    if (d > tv[KNN - 1]) {
      float cv = d; int ci = j;
#pragma unroll
      for (int s = 0; s < KNN; ++s) {
        if (cv > tv[s]) {
          float ov = tv[s]; int oi = ti[s];
          tv[s] = cv; ti[s] = ci; cv = ov; ci = oi;
        }
      }
    }
  }
  int* op = idx + (size_t)(b * NPTS + r) * KNN;
#pragma unroll
  for (int s = 0; s < KNN; ++s) op[s] = ti[s];
}

// ---------------------------------------------------------------------------
// Kernel 3: fused transformer, 16 nodes / workgroup, 128 thr = 4 wave32 waves.
// Weights arrive via one TDM tensor_load_to_lds (102 KB pre-packed blob);
// all GEMMs are v_wmma_f32_16x16x32_f16 with fully vectorized fragment loads.
// ---------------------------------------------------------------------------
__global__ void __launch_bounds__(128)
former_kernel(const float* __restrict__ x0, const int* __restrict__ idx,
              const _Float16* __restrict__ wblob, float* __restrict__ out) {
  __shared__ __align__(64) _Float16 s_wpack[NTILES * 512];  // 102 KB
  __shared__ float s_tgt[16 * 32];        // K-padded (cols 8..31 zero)
  __shared__ float s_tgte[16 * DIM];
  __shared__ float s_q[16 * DIM];
  __shared__ int   s_idx[4][KNN];
  __shared__ float s_sx[4][KNN * 32];     // K-padded (cols 8..31 zero)
  __shared__ float s_se[4][KNN * DIM];
  __shared__ float s_k[4][KNN * DIM];
  __shared__ float s_v[4][KNN * DIM];
  __shared__ float s_sc[4][NH * KNN];
  __shared__ float s_o[16 * DIM];
  __shared__ float s_h1[16 * DIM];
  __shared__ float s_hf[16 * FF];
  __shared__ float s_h2[16 * DIM];

  int tid = threadIdx.x;
  int w = tid >> 5;
  int lane = tid & 31;
  int b = blockIdx.x >> 9;
  int n0 = (blockIdx.x & 511) * 16;
  const float* x0b = x0 + (size_t)b * NPTS * CL;

  // --- TDM: DMA the packed weight blob into LDS (wave 0 issues one op) ---
  if (w == 0) {
    unsigned long long ga = (unsigned long long)(size_t)wblob;
    unsigned int la = (unsigned int)(size_t)(&s_wpack[0]);
    u32x4 g0;
    g0[0] = 1u;                                        // count=1, user desc
    g0[1] = la;                                        // lds_addr
    g0[2] = (unsigned int)ga;                          // global_addr[31:0]
    g0[3] = (unsigned int)(ga >> 32) | (2u << 30);     // addr[56:32] | type=2
    u32x8 g1;
    g1[0] = 2u << 16;                                  // data_size = 4 bytes
    g1[1] = (WBLOB_DW_X & 0xFFFFu) << 16;              // tensor_dim0[15:0]
    g1[2] = WBLOB_DW_Y << 16;                          // tensor_dim1[15:0]
    g1[3] = WBLOB_DW_X << 16;                          // tile_dim0
    g1[4] = WBLOB_DW_Y;                                // tile_dim1 (tile_dim2=0)
    g1[5] = WBLOB_DW_X;                                // tensor_dim0_stride
    g1[6] = 0u;
    g1[7] = 0u;
    asm volatile("tensor_load_to_lds %0, %1" :: "s"(g0), "s"(g1) : "memory");
  }
  // --- zero the K-pad columns (disjoint from the staging writes below) ---
  for (int i = tid; i < 16 * 24; i += 128) {
    int r = i / 24, c = 8 + i % 24;
    s_tgt[r * 32 + c] = 0.0f;
  }
  for (int i = tid; i < 4 * 16 * 24; i += 128) {
    int wv = i / (16 * 24), rr = i % (16 * 24);
    int r = rr / 24, c = 8 + rr % 24;
    s_sx[wv][r * 32 + c] = 0.0f;
  }
  // --- stage tgt rows (16 x 8) while the DMA flies ---
  if (tid < 16 * CL)
    s_tgt[(tid >> 3) * 32 + (tid & 7)] =
        x0b[(size_t)(n0 + (tid >> 3)) * CL + (tid & 7)];
  if (w == 0) __builtin_amdgcn_s_wait_tensorcnt(0);
  __syncthreads();

  // --- tgt_e = tgt @ Wemb : each wave one 16x16 N-tile (K=8 pad 32) ---
  gemm_tile(s_tgt, 32, 32, s_wpack, T_EMB, 4, w * 16, s_tgte, DIM, lane, false);
  __syncthreads();
  // --- q = tgt_e @ Wq ---
  gemm_tile(s_tgte, DIM, DIM, s_wpack, T_WQ, 4, w * 16, s_q, DIM, lane, false);
  __syncthreads();

  // --- per-node attention: wave w handles tile-row i = iter*4 + w ---
  for (int iter = 0; iter < 4; ++iter) {
    int i = iter * 4 + w;
    int node = n0 + i;
    if (lane < KNN)
      s_idx[w][lane] = idx[(size_t)(b * NPTS + node) * KNN + lane];
    __syncthreads();
#pragma unroll
    for (int e = 0; e < 4; ++e) {
      int f = lane * 4 + e;  // 0..127
      int r = f >> 3, c = f & 7;
      s_sx[w][r * 32 + c] = x0b[(size_t)s_idx[w][r] * CL + c];
    }
    __syncthreads();
    // src_e = sx @ Wemb (16x8 @ 8x64, K padded to 32)
    for (int t = 0; t < 4; ++t)
      gemm_tile(s_sx[w], 32, 32, s_wpack, T_EMB, 4, t * 16, s_se[w], DIM, lane, false);
    __syncthreads();
    // k = src_e @ Wk ; v = src_e @ Wv
    for (int t = 0; t < 4; ++t) {
      gemm_tile(s_se[w], DIM, DIM, s_wpack, T_WK, 4, t * 16, s_k[w], DIM, lane, false);
      gemm_tile(s_se[w], DIM, DIM, s_wpack, T_WV, 4, t * 16, s_v[w], DIM, lane, false);
    }
    __syncthreads();
    // scores: att[h][j] = 0.25 * sum_d q[i][h*16+d] * k[j][h*16+d]
    {
      int j = lane & 15;
      int h0 = lane >> 4;
#pragma unroll
      for (int hh = 0; hh < 2; ++hh) {
        int h = h0 + 2 * hh;
        float s = 0.0f;
#pragma unroll
        for (int d = 0; d < DH; ++d)
          s += s_q[i * DIM + h * DH + d] * s_k[w][j * DIM + h * DH + d];
        s_sc[w][h * KNN + j] = s * 0.25f;
      }
    }
    __syncthreads();
    if (lane < NH) {  // softmax per head
      float mx = -1e30f;
#pragma unroll
      for (int j = 0; j < KNN; ++j) mx = fmaxf(mx, s_sc[w][lane * KNN + j]);
      float sum = 0.0f;
#pragma unroll
      for (int j = 0; j < KNN; ++j) {
        float e = __expf(s_sc[w][lane * KNN + j] - mx);
        s_sc[w][lane * KNN + j] = e;
        sum += e;
      }
      float is = 1.0f / sum;
#pragma unroll
      for (int j = 0; j < KNN; ++j) s_sc[w][lane * KNN + j] *= is;
    }
    __syncthreads();
#pragma unroll
    for (int t = 0; t < 2; ++t) {  // o[i][dp] = sum_j p[h][j]*v[j][dp]
      int dp = lane + 32 * t;
      int h = dp >> 4;
      float acc = 0.0f;
#pragma unroll
      for (int j = 0; j < KNN; ++j)
        acc += s_sc[w][h * KNN + j] * s_v[w][j * DIM + dp];
      s_o[i * DIM + dp] = acc;
    }
    __syncthreads();
  }

  // --- att_proj = O @ Wo -> s_h1 (raw) ---
  gemm_tile(s_o, DIM, DIM, s_wpack, T_WO, 4, w * 16, s_h1, DIM, lane, false);
  __syncthreads();
  // --- h1 = LN(tgt_e + att_proj) ---
  if (tid < 16) {
    float s = 0.0f, s2 = 0.0f;
    for (int d = 0; d < DIM; ++d) {
      float x = s_tgte[tid * DIM + d] + s_h1[tid * DIM + d];
      s += x; s2 += x * x;
    }
    float m = s * (1.0f / DIM);
    float var = s2 * (1.0f / DIM) - m * m;
    float ri = rsqrtf(var + 1e-5f);
    for (int d = 0; d < DIM; ++d) {
      float x = s_tgte[tid * DIM + d] + s_h1[tid * DIM + d];
      s_h1[tid * DIM + d] = (x - m) * ri;
    }
  }
  __syncthreads();
  // --- hf = relu(h1 @ Wff1) : 16 N-tiles, 4 per wave ---
  for (int t = 0; t < 4; ++t)
    gemm_tile(s_h1, DIM, DIM, s_wpack, T_FF1, 16, (w * 4 + t) * 16, s_hf, FF, lane, true);
  __syncthreads();
  // --- ff2 = hf @ Wff2 -> s_h2 (raw) ---
  gemm_tile(s_hf, FF, FF, s_wpack, T_FF2, 4, w * 16, s_h2, DIM, lane, false);
  __syncthreads();
  // --- h2 = LN(h1 + ff2) ---
  if (tid < 16) {
    float s = 0.0f, s2 = 0.0f;
    for (int d = 0; d < DIM; ++d) {
      float x = s_h1[tid * DIM + d] + s_h2[tid * DIM + d];
      s += x; s2 += x * x;
    }
    float m = s * (1.0f / DIM);
    float var = s2 * (1.0f / DIM) - m * m;
    float ri = rsqrtf(var + 1e-5f);
    for (int d = 0; d < DIM; ++d) {
      float x = s_h1[tid * DIM + d] + s_h2[tid * DIM + d];
      s_h2[tid * DIM + d] = (x - m) * ri;
    }
  }
  __syncthreads();
  // --- sq_c = h2 @ Wout (64x8, N padded to 16) : wave 0 only ---
  if (w == 0) {
    v8f c = {};
    for (int k0 = 0; k0 < DIM; k0 += 32) {
      v16h a = frag_a(s_h2, DIM, k0, lane);
      v16h bb = frag_b_packed(s_wpack, T_OUT + (k0 >> 5), lane);
      c = __builtin_amdgcn_wmma_f32_16x16x32_f16(false, a, false, bb, (short)0,
                                                 c, false, false);
    }
    int n = lane & 15;
    int mo = (lane >> 4) * 8;
    if (n < CL) {
#pragma unroll
      for (int v = 0; v < 8; ++v)
        out[(size_t)(b * NPTS + n0 + v + mo) * CL + n] = c[v];
    }
  }
}

// ---------------------------------------------------------------------------
extern "C" void kernel_launch(void* const* d_in, const int* in_sizes, int n_in,
                              void* d_out, int out_size, void* d_ws,
                              size_t ws_size, hipStream_t stream) {
  const float* xc   = (const float*)d_in[0];
  const float* Wemb = (const float*)d_in[1];
  const float* Wqkv = (const float*)d_in[2];
  const float* Wo   = (const float*)d_in[3];
  const float* Wff1 = (const float*)d_in[4];
  const float* Wff2 = (const float*)d_in[5];
  const float* Wout = (const float*)d_in[6];
  float* out = (float*)d_out;

  float* x0w = (float*)d_ws;                           // 1 MB
  float* xnw = x0w + (size_t)BSZ * NPTS * CL;          // 1 MB
  int* idxw  = (int*)(xnw + (size_t)BSZ * NPTS * CL);  // 2 MB
  _Float16* blob = (_Float16*)((char*)d_ws + (size_t)4 * 1024 * 1024);  // 102 KB

  prep_kernel<<<(BSZ * NPTS) / 256, 256, 0, stream>>>(xc, x0w, xnw, out);
  pack_kernel<<<NTILES, 256, 0, stream>>>(Wemb, Wqkv, Wo, Wff1, Wff2, Wout, blob);
  topk_kernel<<<BSZ * (NPTS / 256), 256, 0, stream>>>(xnw, idxw);
  former_kernel<<<BSZ * (NPTS / 16), 128, 0, stream>>>(x0w, idxw, blob, out);
}
